// TransformerTowerBlock_22325240005292
// MI455X (gfx1250) — matrix operand order repaired and checked by reading, not compiled, not run on hardware
//
#include <hip/hip_runtime.h>
#include <math.h>

typedef __bf16 bf16_t;
typedef __bf16 v16bf __attribute__((ext_vector_type(16)));
typedef float  v8f   __attribute__((ext_vector_type(8)));
typedef int    v4i_t __attribute__((ext_vector_type(4)));

#define EPSV 1e-5f

// ---------------- CDNA5 async global->LDS copy ----------------
#if __has_builtin(__builtin_amdgcn_global_load_async_to_lds_b128)
#define HAVE_ASYNC_BUILTIN 1
#endif

__device__ __forceinline__ void async_b128(const bf16_t* g, bf16_t* l) {
#ifdef HAVE_ASYNC_BUILTIN
  __builtin_amdgcn_global_load_async_to_lds_b128(
      (__attribute__((address_space(1))) v4i_t*)(size_t)g,
      (__attribute__((address_space(3))) v4i_t*)(void*)l,
      0, 0);
#else
  unsigned lo = (unsigned)(size_t)(__attribute__((address_space(3))) void*)l;
  asm volatile("global_load_async_to_lds_b128 %0, %1, off" :: "v"(lo), "v"(g) : "memory");
#endif
}

#if __has_builtin(__builtin_amdgcn_s_wait_asynccnt)
#define WAIT_ASYNC(n) __builtin_amdgcn_s_wait_asynccnt(n)
#else
#define WAIT_ASYNC(n) asm volatile("s_wait_asynccnt %0" :: "i"(n) : "memory")
#endif

// Hard scheduling fence: keep fragment-load groups from sinking into WMMA chains.
#if __has_builtin(__builtin_amdgcn_sched_barrier)
#define SCHED_FENCE() __builtin_amdgcn_sched_barrier(0)
#else
#define SCHED_FENCE() asm volatile("" ::: "memory")
#endif

// ---------------- WMMA helpers ----------------
union FragU { uint4 u[2]; v16bf v; };

// Load a 16x32 bf16 A/B fragment from a row-major [rows][stride] bf16 tile.
// ISA layout (16-bit A 16x32): lane L (m = L&15, half = L>>4) holds
// K = {half*8..half*8+7, 16+half*8..16+half*8+7}.
__device__ __forceinline__ v16bf frag_ld(const bf16_t* base, int row0, int stride, int k0) {
  int lane = threadIdx.x & 31;
  int m = lane & 15, half = lane >> 4;
  const uint4* p = (const uint4*)(base + (size_t)(row0 + m) * stride + k0);
  FragU f; f.u[0] = p[half]; f.u[1] = p[half + 2];
  return f.v;
}

__device__ __forceinline__ v8f wmma_bf16(v16bf a, v16bf b, v8f c) {
  return __builtin_amdgcn_wmma_f32_16x16x32_bf16(false, a, false, b, (short)0, c, false, false);
}

// ---------------- small kernels ----------------

// per-channel mean of squares over rows: in [R][C] -> ms[C]
__global__ void colms_kernel(const float* __restrict__ in, float* __restrict__ ms, int R, int C) {
  int c = blockIdx.x;
  float acc = 0.f;
  for (int r = threadIdx.x; r < R; r += 256) { float v = in[(size_t)r * C + c]; acc += v * v; }
  __shared__ float red[256];
  red[threadIdx.x] = acc; __syncthreads();
  for (int s = 128; s > 0; s >>= 1) { if (threadIdx.x < s) red[threadIdx.x] += red[threadIdx.x + s]; __syncthreads(); }
  if (threadIdx.x == 0) ms[c] = red[0] / (float)R;
}

// pair bias: pb[b,i,j,g] = sum_c gelu(pair*rsqrt(msp)*g_bn) * w[g][c]
__global__ void pb_kernel(const float* __restrict__ pair, const float* __restrict__ msp,
                          const float* __restrict__ abg, const float* __restrict__ w,
                          float* __restrict__ pb) {
  int bi = blockIdx.x;            // b*128+i
  int j = threadIdx.x;            // 0..127
  const float* row = pair + ((size_t)bi * 128 + j) * 128;
  float s0 = 0.f, s1 = 0.f;
  for (int c = 0; c < 128; c++) {
    float t = row[c] * rsqrtf(msp[c] + EPSV) * abg[c];
    float ge = 0.5f * t * (1.f + erff(t * 0.70710678118654752f));
    s0 += ge * w[c]; s1 += ge * w[128 + c];
  }
  float* o = pb + ((size_t)bi * 128 + j) * 2;
  o[0] = s0; o[1] = s1;
}

__global__ void f2bf_kernel(const float* __restrict__ in, bf16_t* __restrict__ out, long n) {
  long i = (long)blockIdx.x * 256 + threadIdx.x;
  if (i < n) out[i] = (bf16_t)in[i];
}

// out_bf = bf16(x * rsqrt(ms[c]) * g[c])
__global__ void scalebf_kernel(const float* __restrict__ x, const float* __restrict__ ms,
                               const float* __restrict__ g, bf16_t* __restrict__ out, long n, int C) {
  long i = (long)blockIdx.x * 256 + threadIdx.x;
  if (i < n) { int c = (int)(i % C); out[i] = (bf16_t)(x[i] * rsqrtf(ms[c] + EPSV) * g[c]); }
}

// out = x + t * rsqrt(ms[c]) * g[c]
__global__ void resid_kernel(const float* __restrict__ x, const float* __restrict__ t,
                             const float* __restrict__ ms, const float* __restrict__ g,
                             float* __restrict__ out, long n, int C) {
  long i = (long)blockIdx.x * 256 + threadIdx.x;
  if (i < n) { int c = (int)(i % C); out[i] = x[i] + t[i] * rsqrtf(ms[c] + EPSV) * g[c]; }
}

// per-(token,head) LayerNorm over 128 ch + optional AlphaGenome RoPE; writes bf16
__global__ void lnrope_kernel(const float* __restrict__ src, const float* __restrict__ g,
                              const float* __restrict__ bvec, bf16_t* __restrict__ dst,
                              int H, int S, int do_rope) {
  int idx = blockIdx.x;           // bs*H + h
  int t = threadIdx.x;            // 0..127
  int bs = idx / H;
  const float* p = src + (size_t)idx * 128;
  __shared__ float sh[128];
  float v = p[t];
  sh[t] = v; __syncthreads();
  for (int s2 = 64; s2 > 0; s2 >>= 1) { if (t < s2) sh[t] += sh[t + s2]; __syncthreads(); }
  float mean = sh[0] * (1.f / 128.f); __syncthreads();
  float d = v - mean;
  sh[t] = d * d; __syncthreads();
  for (int s2 = 64; s2 > 0; s2 >>= 1) { if (t < s2) sh[t] += sh[t + s2]; __syncthreads(); }
  float var = sh[0] * (1.f / 128.f); __syncthreads();
  float y = d * rsqrtf(var + EPSV) * g[t] + bvec[t];
  float out = y;
  if (do_rope) {
    sh[t] = y; __syncthreads();
    int i = t >> 1;
    // inv_freq = 1/(i + 1985^(i/63)), theta = pos * inv_freq (pairs share theta)
    float invf = 1.f / ((float)i + __expf(((float)i * (1.f / 63.f)) * logf(1985.f)));
    float th = (float)(bs % S) * invf;
    float c = cosf(th), s = sinf(th);
    float partner = (t & 1) ? sh[t - 1] : sh[t + 1];
    out = (t & 1) ? (y * c + partner * s) : (y * c - partner * s);
  }
  dst[(size_t)idx * 128 + t] = (bf16_t)out;
}

// ---------------- WMMA GEMM: C[M,N] = A[M,K] * W[N,K]^T (+bias)(+relu) ----------------
// Double-buffered async global->LDS pipeline, BK=64; fragment loads fenced so
// each group of 8 WMMAs issues back-to-back while the next group's loads overlap.
// mode 0: store f32; mode 1: +bias store f32; mode 2: +bias, relu, store bf16
__global__ __launch_bounds__(256) void gemm_bf16_kernel(
    const bf16_t* __restrict__ A, const bf16_t* __restrict__ W,
    const float* __restrict__ bias, float* __restrict__ Cf, bf16_t* __restrict__ Cb,
    int M, int N, int K, int mode) {
  extern __shared__ char gsm[];
  bf16_t* a_t = (bf16_t*)gsm;               // 2 bufs * 128*64
  bf16_t* w_t = a_t + 2 * 128 * 64;         // 2 bufs * 128*64
  int m0 = blockIdx.x * 128, n0 = blockIdx.y * 128;
  int tid = threadIdx.x, wave = tid >> 5;
  v8f acc[8];
  v8f zero = {0.f, 0.f, 0.f, 0.f, 0.f, 0.f, 0.f, 0.f};
#pragma unroll
  for (int i = 0; i < 8; i++) acc[i] = zero;

  int nk = K >> 6;                          // K is a multiple of 64 for all layers
  auto issue = [&](int kt, int buf) {
    int kb = kt * 64;
    bf16_t* ab = a_t + buf * (128 * 64);
    bf16_t* wb = w_t + buf * (128 * 64);
#pragma unroll
    for (int j = 0; j < 4; j++) {
      int idx = tid + j * 256;              // 1024 x 16B chunks per tile
      int row = idx >> 3, col = idx & 7;    // 8 chunks per 64-wide row
      async_b128(A + (size_t)(m0 + row) * K + kb + col * 8, ab + idx * 8);
      async_b128(W + (size_t)(n0 + row) * K + kb + col * 8, wb + idx * 8);
    }
  };

  issue(0, 0);
  int cur = 0;
  for (int kt = 0; kt < nk; kt++) {
    if (kt + 1 < nk) { issue(kt + 1, cur ^ 1); WAIT_ASYNC(8); }  // 8 newer in flight
    else             { WAIT_ASYNC(0); }
    __syncthreads();                         // all waves' current tile arrived
    const bf16_t* ab = a_t + cur * (128 * 64);
    const bf16_t* wb = w_t + cur * (128 * 64);
    // group 1: batch loads, fence, then 8 WMMAs back-to-back
    v16bf af0 = frag_ld(ab, wave * 16, 64, 0);
    v16bf bfr0[8];
#pragma unroll
    for (int nt = 0; nt < 8; nt++) bfr0[nt] = frag_ld(wb, nt * 16, 64, 0);
    SCHED_FENCE();
#pragma unroll
    for (int nt = 0; nt < 8; nt++) acc[nt] = wmma_bf16(af0, bfr0[nt], acc[nt]);
    // group 2 loads live in the same region as group-1 WMMAs -> may overlap them
    v16bf af1 = frag_ld(ab, wave * 16, 64, 32);
    v16bf bfr1[8];
#pragma unroll
    for (int nt = 0; nt < 8; nt++) bfr1[nt] = frag_ld(wb, nt * 16, 64, 32);
    SCHED_FENCE();
#pragma unroll
    for (int nt = 0; nt < 8; nt++) acc[nt] = wmma_bf16(af1, bfr1[nt], acc[nt]);
    __syncthreads();                         // everyone done reading buf[cur]
    cur ^= 1;
  }

  int lane = tid & 31, half = lane >> 4, nlo = lane & 15;
  for (int nt = 0; nt < 8; nt++) {
    int n = n0 + nt * 16 + nlo;
    float bv = bias ? bias[n] : 0.f;
#pragma unroll
    for (int r = 0; r < 8; r++) {
      int m = m0 + wave * 16 + r + 8 * half;
      float v = acc[nt][r] + bv;
      if (mode == 2) { v = v > 0.f ? v : 0.f; Cb[(size_t)m * N + n] = (bf16_t)v; }
      else           { Cf[(size_t)m * N + n] = v; }
    }
  }
}

// ---------------- Flash attention with tanh soft-clip + tile-constant bias ----------------
// Q:[B,S,8,128] bf16, K/V:[B,S,4,128] bf16, pb:[B,128,128,2] f32, Y:[B,8,S,128] bf16
__global__ __launch_bounds__(256) void attn_kernel(
    const bf16_t* __restrict__ Q, const bf16_t* __restrict__ Kb, const bf16_t* __restrict__ Vb,
    const float* __restrict__ pbias, bf16_t* __restrict__ Y) {
  extern __shared__ char smem[];
  bf16_t* q_lds  = (bf16_t*)smem;            // 128x128
  bf16_t* k_lds  = q_lds + 128 * 128;        // 64x128
  bf16_t* vt_lds = k_lds + 64 * 128;         // 128x64 (transposed V)
  bf16_t* p_lds  = vt_lds + 128 * 64;        // 8 waves * 16x64

  int bid = blockIdx.x;
  int qb = bid & 15, h = (bid >> 4) & 7, b = bid >> 7;
  int kvh = h & 3, g = h >> 2;
  int q0 = qb * 128;
  int tid = threadIdx.x, wave = tid >> 5, lane = tid & 31, half = lane >> 4, nlo = lane & 15;

  { // Q tile via async copy: rows q0..q0+127, global row stride 8*128 bf16
    const bf16_t* qbase = Q + ((size_t)(b * 2048 + q0) * 8 + h) * 128;
#pragma unroll
    for (int j = 0; j < 8; j++) {
      int idx = tid + j * 256;               // 2048 x 16B chunks
      int row = idx >> 4, col = idx & 15;
      async_b128(qbase + (size_t)row * 1024 + col * 8, q_lds + idx * 8);
    }
  }

  float mrun[8], lrun[8];
  v8f O[8];
  v8f zero = {0.f, 0.f, 0.f, 0.f, 0.f, 0.f, 0.f, 0.f};
#pragma unroll
  for (int r = 0; r < 8; r++) { mrun[r] = -1e30f; lrun[r] = 0.f; }
#pragma unroll
  for (int i = 0; i < 8; i++) O[i] = zero;
  WAIT_ASYNC(0);
  __syncthreads();

  // Q fragments are invariant over all key blocks: hoist into registers.
  v16bf qf[4];
#pragma unroll
  for (int ks = 0; ks < 4; ks++) qf[ks] = frag_ld(q_lds, wave * 16, 128, ks * 32);

  const float scale = 0.088388347648318447f;  // 1/sqrt(128)
  bf16_t* pw = p_lds + wave * 16 * 64;

  for (int kb = 0; kb < 32; kb++) {           // 32 key blocks of 64
    int t0 = kb * 64;
    { // K tile 32x128 (rows 0..31) async (natural [t][d]); V transposed manually to vt[d][t]
      const bf16_t* kbase = Kb + ((size_t)(b * 2048 + t0) * 4 + kvh) * 128;
#pragma unroll
      for (int j = 0; j < 2; j++) {
        int idx = tid + j * 256;              // 512 x 16B chunks
        int row = idx >> 4, col = idx & 15;
        async_b128(kbase + (size_t)row * 512 + col * 8, k_lds + idx * 8);
      }
      const bf16_t* vbase = Vb + ((size_t)(b * 2048 + t0) * 4 + kvh) * 128;
#pragma unroll
      for (int j = 0; j < 4; j++) {
        int idx = tid + j * 256;              // 1024 x 8-elem chunks
        int trow = idx >> 4, c8 = idx & 15;   // 16 chunks per 128-wide row
        union { uint4 u; bf16_t e[8]; } tmp;
        tmp.u = *((const uint4*)(vbase + (size_t)trow * 512) + c8);
        int d0 = c8 * 8;
#pragma unroll
        for (int e = 0; e < 8; e++) vt_lds[(d0 + e) * 64 + trow] = tmp.e[e];
      }
    }
    WAIT_ASYNC(0);
    __syncthreads();

    // S = Q K^T  (16 rows per wave x 64 keys); fenced load/WMMA groups
    v8f Sa[4];
#pragma unroll
    for (int i = 0; i < 4; i++) Sa[i] = zero;
#pragma unroll
    for (int ks = 0; ks < 4; ks++) {
      v16bf kf[4];
#pragma unroll
      for (int nt = 0; nt < 4; nt++) kf[nt] = frag_ld(k_lds, nt * 16, 128, ks * 32);
      SCHED_FENCE();
#pragma unroll
      for (int nt = 0; nt < 4; nt++) Sa[nt] = wmma_bf16(qf[ks], kf[nt], Sa[nt]);
    }

    // bias is constant over each 16x16 tile (x16 upsample of pb)
    float biasv[4];
#pragma unroll
    for (int nt = 0; nt < 4; nt++) {
      int q16 = qb * 8 + wave;
      int t16 = kb * 4 + nt;
      biasv[nt] = pbias[((size_t)(b * 128 + q16) * 128 + t16) * 2 + g];
    }

    // tanh clip + online softmax (row-wise, wave-local)
#pragma unroll
    for (int r = 0; r < 8; r++) {
      float sv[4];
      float mx = -1e30f;
#pragma unroll
      for (int nt = 0; nt < 4; nt++) {
        float s = Sa[nt][r] * scale + biasv[nt];
        s = 5.f * tanhf(s * 0.2f);
        sv[nt] = s; mx = fmaxf(mx, s);
      }
#pragma unroll
      for (int off = 8; off > 0; off >>= 1) mx = fmaxf(mx, __shfl_xor(mx, off, 32));
      float mnew = fmaxf(mrun[r], mx);
      float alpha = __expf(mrun[r] - mnew);
      mrun[r] = mnew;
      lrun[r] *= alpha;
#pragma unroll
      for (int dt = 0; dt < 8; dt++) O[dt][r] *= alpha;
      float rs = 0.f;
#pragma unroll
      for (int nt = 0; nt < 4; nt++) {
        float pexp = __expf(sv[nt] - mnew);
        rs += pexp;
        pw[(r + 8 * half) * 64 + nt * 16 + nlo] = (bf16_t)pexp;  // D-layout -> [m][t]
      }
#pragma unroll
      for (int off = 8; off > 0; off >>= 1) rs += __shfl_xor(rs, off, 32);
      lrun[r] += rs;
    }
    asm volatile("s_wait_dscnt 0" ::: "memory");  // P staged in wave-private LDS

    // O += P * V   (A = P 16x64, B from vt[d][t]); fenced load/WMMA groups
#pragma unroll
    for (int ks = 0; ks < 2; ks++) {
      v16bf pf = frag_ld(pw, 0, 64, ks * 32);
      v16bf vf[8];
#pragma unroll
      for (int dt = 0; dt < 8; dt++) vf[dt] = frag_ld(vt_lds, dt * 16, 64, ks * 32);
      SCHED_FENCE();
#pragma unroll
      for (int dt = 0; dt < 8; dt++) O[dt] = wmma_bf16(pf, vf[dt], O[dt]);
    }
    __syncthreads();
  }

  // normalize and store in scrambled [B,H,S,D] order (matches reference permute+reshape)
#pragma unroll
  for (int r = 0; r < 8; r++) {
    float inv = 1.f / lrun[r];
    int srow = q0 + wave * 16 + r + 8 * half;
    size_t base = (((size_t)b * 8 + h) * 2048 + srow) * 128;
#pragma unroll
    for (int dt = 0; dt < 8; dt++) Y[base + dt * 16 + nlo] = (bf16_t)(O[dt][r] * inv);
  }
}

// ---------------- orchestration ----------------
extern "C" void kernel_launch(void* const* d_in, const int* in_sizes, int n_in,
                              void* d_out, int out_size, void* d_ws, size_t ws_size,
                              hipStream_t stream) {
  const float* x      = (const float*)d_in[0];
  const float* pair   = (const float*)d_in[1];
  const float* ab_bn_g= (const float*)d_in[2];
  const float* ab_fc_w= (const float*)d_in[3];
  const float* bn1_g  = (const float*)d_in[4];
  const float* wq     = (const float*)d_in[5];
  const float* wk     = (const float*)d_in[6];
  const float* wv     = (const float*)d_in[7];
  const float* qn_g   = (const float*)d_in[8];
  const float* qn_b   = (const float*)d_in[9];
  const float* kn_g   = (const float*)d_in[10];
  const float* kn_b   = (const float*)d_in[11];
  const float* vn_g   = (const float*)d_in[12];
  const float* vn_b   = (const float*)d_in[13];
  const float* wo     = (const float*)d_in[14];
  const float* bo     = (const float*)d_in[15];
  const float* bn2_g  = (const float*)d_in[16];
  const float* mbn1_g = (const float*)d_in[17];
  const float* w1     = (const float*)d_in[18];
  const float* b1     = (const float*)d_in[19];
  const float* w2     = (const float*)d_in[20];
  const float* b2     = (const float*)d_in[21];
  const float* mbn2_g = (const float*)d_in[22];

  const int Bv = 2, S = 2048, C = 768, M = Bv * S;   // 4096
  const long nx = (long)M * C;                       // 3,145,728

  char* ws = (char*)d_ws;
  size_t off = 0;
  auto alloc = [&](size_t bytes) -> char* {
    char* p = ws + off;
    off += (bytes + 255) & ~(size_t)255;
    return p;
  };
  float* ms1   = (float*)alloc(768 * 4);
  float* msp   = (float*)alloc(128 * 4);
  float* ms2   = (float*)alloc(768 * 4);
  float* ms3   = (float*)alloc(768 * 4);
  float* ms4   = (float*)alloc(768 * 4);
  float* pb    = (float*)alloc((size_t)2 * 128 * 128 * 2 * 4);
  bf16_t* wq_bf = (bf16_t*)alloc((size_t)1024 * 768 * 2);
  bf16_t* wk_bf = (bf16_t*)alloc((size_t)512 * 768 * 2);
  bf16_t* wv_bf = (bf16_t*)alloc((size_t)512 * 768 * 2);
  bf16_t* wo_bf = (bf16_t*)alloc((size_t)768 * 1024 * 2);
  bf16_t* w1_bf = (bf16_t*)alloc((size_t)1536 * 768 * 2);
  bf16_t* w2_bf = (bf16_t*)alloc((size_t)768 * 1536 * 2);
  bf16_t* xn_bf = (bf16_t*)alloc((size_t)M * 768 * 2);
  float* q_pre  = (float*)alloc((size_t)M * 1024 * 4);
  float* k_pre  = (float*)alloc((size_t)M * 512 * 4);
  float* v_pre  = (float*)alloc((size_t)M * 512 * 4);
  bf16_t* q_bf  = (bf16_t*)alloc((size_t)M * 1024 * 2);
  bf16_t* k_bf  = (bf16_t*)alloc((size_t)M * 512 * 2);
  bf16_t* v_bf  = (bf16_t*)alloc((size_t)M * 512 * 2);
  bf16_t* y_bf  = (bf16_t*)alloc((size_t)M * 1024 * 2);
  float* t1     = (float*)alloc((size_t)M * 768 * 4);
  float* x2     = (float*)alloc((size_t)M * 768 * 4);
  bf16_t* xm_bf = (bf16_t*)alloc((size_t)M * 768 * 2);
  bf16_t* h_bf  = (bf16_t*)alloc((size_t)M * 1536 * 2);
  float* t2     = (float*)alloc((size_t)M * 768 * 4);

  // --- stats for rms_bn(x) and rms_bn(pair) ---
  colms_kernel<<<768, 256, 0, stream>>>(x, ms1, M, 768);
  colms_kernel<<<128, 256, 0, stream>>>(pair, msp, 2 * 128 * 128, 128);

  // --- pair bias ---
  pb_kernel<<<2 * 128, 128, 0, stream>>>(pair, msp, ab_bn_g, ab_fc_w, pb);

  // --- bf16 weight conversions ---
  f2bf_kernel<<<(1024 * 768 + 255) / 256, 256, 0, stream>>>(wq, wq_bf, (long)1024 * 768);
  f2bf_kernel<<<(512 * 768 + 255) / 256, 256, 0, stream>>>(wk, wk_bf, (long)512 * 768);
  f2bf_kernel<<<(512 * 768 + 255) / 256, 256, 0, stream>>>(wv, wv_bf, (long)512 * 768);
  f2bf_kernel<<<(768 * 1024 + 255) / 256, 256, 0, stream>>>(wo, wo_bf, (long)768 * 1024);
  f2bf_kernel<<<(1536 * 768 + 255) / 256, 256, 0, stream>>>(w1, w1_bf, (long)1536 * 768);
  f2bf_kernel<<<(768 * 1536 + 255) / 256, 256, 0, stream>>>(w2, w2_bf, (long)768 * 1536);

  // --- xn = rms_bn(x) in bf16 ---
  scalebf_kernel<<<(int)((nx + 255) / 256), 256, 0, stream>>>(x, ms1, bn1_g, xn_bf, nx, 768);

  const size_t gemm_lds = (size_t)4 * 128 * 64 * sizeof(bf16_t);  // 64KB (2x double-buffered tiles)

  // --- QKV projections ---
  gemm_bf16_kernel<<<dim3(M / 128, 1024 / 128), 256, gemm_lds, stream>>>(xn_bf, wq_bf, nullptr, q_pre, nullptr, M, 1024, 768, 0);
  gemm_bf16_kernel<<<dim3(M / 128, 512 / 128), 256, gemm_lds, stream>>>(xn_bf, wk_bf, nullptr, k_pre, nullptr, M, 512, 768, 0);
  gemm_bf16_kernel<<<dim3(M / 128, 512 / 128), 256, gemm_lds, stream>>>(xn_bf, wv_bf, nullptr, v_pre, nullptr, M, 512, 768, 0);

  // --- per-head LN (+RoPE for q,k) -> bf16 ---
  lnrope_kernel<<<M * 8, 128, 0, stream>>>(q_pre, qn_g, qn_b, q_bf, 8, S, 1);
  lnrope_kernel<<<M * 4, 128, 0, stream>>>(k_pre, kn_g, kn_b, k_bf, 4, S, 1);
  lnrope_kernel<<<M * 4, 128, 0, stream>>>(v_pre, vn_g, vn_b, v_bf, 4, S, 0);

  // --- flash attention ---
  size_t attn_lds = (size_t)(128 * 128 + 64 * 128 + 128 * 64 + 8 * 16 * 64) * sizeof(bf16_t); // 80KB
  attn_kernel<<<2 * 8 * 16, 256, attn_lds, stream>>>(q_bf, k_bf, v_bf, pb, y_bf);

  // --- output projection + bo ---
  gemm_bf16_kernel<<<dim3(M / 128, 768 / 128), 256, gemm_lds, stream>>>(y_bf, wo_bf, bo, t1, nullptr, M, 768, 1024, 1);

  // --- x2 = x + rms_bn(t1) ---
  colms_kernel<<<768, 256, 0, stream>>>(t1, ms2, M, 768);
  resid_kernel<<<(int)((nx + 255) / 256), 256, 0, stream>>>(x, t1, ms2, bn2_g, x2, nx, 768);

  // --- MLP ---
  colms_kernel<<<768, 256, 0, stream>>>(x2, ms3, M, 768);
  scalebf_kernel<<<(int)((nx + 255) / 256), 256, 0, stream>>>(x2, ms3, mbn1_g, xm_bf, nx, 768);
  gemm_bf16_kernel<<<dim3(M / 128, 1536 / 128), 256, gemm_lds, stream>>>(xm_bf, w1_bf, b1, nullptr, h_bf, M, 1536, 768, 2);
  gemm_bf16_kernel<<<dim3(M / 128, 768 / 128), 256, gemm_lds, stream>>>(h_bf, w2_bf, b2, t2, nullptr, M, 768, 1536, 1);
  colms_kernel<<<768, 256, 0, stream>>>(t2, ms4, M, 768);
  resid_kernel<<<(int)((nx + 255) / 256), 256, 0, stream>>>(x2, t2, ms4, mbn2_g, (float*)d_out, nx, 768);

  // --- second output: pair_x passthrough ---
  (void)hipMemcpyAsync((float*)d_out + nx, pair, (size_t)2 * 128 * 128 * 128 * sizeof(float),
                       hipMemcpyDeviceToDevice, stream);
}
